// MultiHeadedAttention_6622839571004
// MI455X (gfx1250) — matrix operand order repaired and checked by reading, not compiled, not run on hardware
//
#include <hip/hip_runtime.h>

// ---------------------------------------------------------------------------
// MultiHeadedAttention top-k index pipeline for MI455X (gfx1250, wave32, WMMA)
// B=512, S=256, D=768, K=8, HIST=256
// Output: s_indices [B,8] written as floats to d_out.
// ---------------------------------------------------------------------------

#define BDIM   512
#define SDIM   256
#define DDIM   768
#define KSEL   8
#define HIST   256
#define BS     (BDIM * SDIM)   // 131072 rows total

typedef __bf16 bf16_t;
typedef bf16_t v16bf __attribute__((ext_vector_type(16)));
typedef bf16_t v8bf  __attribute__((ext_vector_type(8)));
typedef float  v8f   __attribute__((ext_vector_type(8)));

#define NEG_INF (-__builtin_inff())

// --- WMMA fragment loaders (16-bit dense layouts, ISA 7.12.2) ---------------
// A matrix 16x32 (MxK): lanes 0-15 -> M=lane, K chunks {0..7},{16..23};
//                       lanes 16-31 -> M=lane-16, K chunks {8..15},{24..31}.
__device__ __forceinline__ v16bf load_a_frag(const bf16_t* base, int ld) {
  const int lane = threadIdx.x & 31;
  const int m    = lane & 15;
  const int koff = (lane >> 4) * 8;
  const bf16_t* p = base + m * ld + koff;
  v8bf lo = *(const v8bf*)(p);
  v8bf hi = *(const v8bf*)(p + 16);
  v16bf a;
#pragma unroll
  for (int i = 0; i < 8; ++i) { a[i] = lo[i]; a[i + 8] = hi[i]; }
  return a;
}

// B matrix 32x16 (KxN), source stored row-major [N, K] (nn.Linear weight /
// key rows): lanes 0-15 -> N=lane, K chunks {0..7},{16..23}; lanes 16-31 -> N=lane-16, K chunks {8..15},{24..31}.
__device__ __forceinline__ v16bf load_b_frag(const bf16_t* base, int ld) {
  const int lane = threadIdx.x & 31;
  const int n    = lane & 15;
  const int kb   = (lane >> 4) * 8;
  const bf16_t* p = base + n * ld + kb;
  v8bf x0 = *(const v8bf*)(p);
  v8bf x1 = *(const v8bf*)(p + 16);
  v16bf b;
#pragma unroll
  for (int i = 0; i < 8; ++i) { b[i] = x0[i]; b[i + 8] = x1[i]; }
  return b;
}

__device__ __forceinline__ v8f wmma_bf16(v16bf a, v16bf b, v8f c) {
  return __builtin_amdgcn_wmma_f32_16x16x32_bf16(
      /*neg_a=*/false, a, /*neg_b=*/false, b,
      /*c_mod=*/(short)0, c, /*reuse_a=*/false, /*reuse_b=*/false);
}

// Async global->LDS copy (CDNA5, ASYNCcnt-tracked). GV mode: 64-bit vaddr.
__device__ __forceinline__ void async_copy_b128(const void* gptr, void* lds_generic) {
  const unsigned lds_off = (unsigned)(unsigned long long)lds_generic; // low 32b = LDS byte offset
  asm volatile("global_load_async_to_lds_b128 %0, %1, off"
               :: "v"(lds_off), "v"(gptr)
               : "memory");
}

// ---------------------------------------------------------------------------
// Kernel 0: fold 1/sqrt(D) into Wq/bq, convert weights to bf16, zero histogram
// ---------------------------------------------------------------------------
__global__ void prep_kernel(const float* __restrict__ Wq,
                            const float* __restrict__ Wk,
                            const float* __restrict__ bq,
                            bf16_t* __restrict__ Whq,
                            bf16_t* __restrict__ Whk,
                            float* __restrict__ bqs,
                            unsigned int* __restrict__ count) {
  const int i = blockIdx.x * blockDim.x + threadIdx.x;
  const float qs = 0.03608439182435161f;  // 1/sqrt(768)
  if (i < DDIM * DDIM) {
    Whq[i] = (bf16_t)(Wq[i] * qs);
    Whk[i] = (bf16_t)(Wk[i]);
  }
  if (i < DDIM) bqs[i] = bq[i] * qs;
  if (i < BDIM * HIST) count[i] = 0u;
}

// ---------------------------------------------------------------------------
// Kernel 1: projection GEMM  out[m,n] = x[m,:] . W[n,:] + bias[n]   (bf16 out)
// Block: 256 threads (8 waves). Block tile: 64 rows x 768 cols.
// x row-strip staged once in LDS as bf16; W read from global (L2-resident).
// ---------------------------------------------------------------------------
__global__ __launch_bounds__(256) void proj_kernel(
    const float* __restrict__ x, const bf16_t* __restrict__ W,
    const float* __restrict__ bias, bf16_t* __restrict__ out) {
  __shared__ bf16_t As[64][DDIM];  // 96 KB

  const int tid = threadIdx.x;
  const int m0  = blockIdx.x * 64;

  // Stage + convert: 64*768 f32 = 12288 float4, 48 per thread, coalesced.
  const float4* xsrc = (const float4*)(x + (size_t)m0 * DDIM);
  for (int i = 0; i < 48; ++i) {
    const int idx = tid + i * 256;
    const int row = idx / 192;          // 192 float4 per row
    const int c4  = idx % 192;
    const float4 v = xsrc[(size_t)row * 192 + c4];
    const int c = c4 * 4;
    As[row][c + 0] = (bf16_t)v.x;
    As[row][c + 1] = (bf16_t)v.y;
    As[row][c + 2] = (bf16_t)v.z;
    As[row][c + 3] = (bf16_t)v.w;
  }
  __syncthreads();

  const int wave   = tid >> 5;
  const int lane   = tid & 31;
  const int mstrip = wave & 3;   // 16-row strip
  const int nhalf  = wave >> 2;  // 384-col half

  for (int ng = 0; ng < 6; ++ng) {
    const int n0 = nhalf * 384 + ng * 64;
    v8f acc[4];
#pragma unroll
    for (int t = 0; t < 4; ++t)
#pragma unroll
      for (int e = 0; e < 8; ++e) acc[t][e] = 0.0f;

    for (int k0 = 0; k0 < DDIM; k0 += 32) {
      const v16bf a = load_a_frag(&As[mstrip * 16][k0], DDIM);
#pragma unroll
      for (int t = 0; t < 4; ++t) {
        const v16bf b = load_b_frag(W + (size_t)(n0 + t * 16) * DDIM + k0, DDIM);
        acc[t] = wmma_bf16(a, b, acc[t]);
      }
    }

    // Epilogue: C layout — lanes 0-15: M=r, lanes 16-31: M=8+r (per VGPR r).
    const int mbase = m0 + mstrip * 16 + (lane >> 4) * 8;
#pragma unroll
    for (int t = 0; t < 4; ++t) {
      const int n = n0 + (lane & 15);
      const float bb = bias[n];
#pragma unroll
      for (int r = 0; r < 8; ++r)
        out[(size_t)(mbase + r) * DDIM + n] = (bf16_t)(acc[t][r] + bb);
    }
  }
}

// ---------------------------------------------------------------------------
// Kernel 2: scores = qh @ kh^T per batch, masked row top-8, histogram.
// Grid: (512 batches, 2 q-halves). Block: 256 threads (8 waves).
// Each wave: 16 q rows x all 256 key columns (16 accumulator tiles).
// kh K-slices (256x32 bf16, 16 KB) are staged into LDS with async
// global->LDS copies, double-buffered on ASYNCcnt, so each slice is read
// from L2 once per block instead of once per wave (8x on-chip traffic cut).
// Softmax skipped (monotone -> identical top-k indices).
// ---------------------------------------------------------------------------
__global__ __launch_bounds__(256) void scores_topk_kernel(
    const bf16_t* __restrict__ qh, const bf16_t* __restrict__ kh,
    const unsigned char* __restrict__ mask, unsigned int* __restrict__ count) {
  // Phase 1 (WMMA): Ks[2][256][32] bf16 double buffer (32 KB) at offset 0.
  // Phase 2 (top-k): sc[8][16][257] f32 (131.5 KB) over the same pool.
  // Phases are separated by barriers, so aliasing is safe.
  __shared__ __align__(16) char smem[8 * 16 * 257 * 4];
  bf16_t (*Ks)[SDIM][32] = (bf16_t (*)[SDIM][32])smem;
  float  (*sc)[16][257]  = (float (*)[16][257])smem;

  const int b     = blockIdx.x;
  const int qhalf = blockIdx.y;
  const int tid   = threadIdx.x;
  const int wave  = tid >> 5;
  const int lane  = tid & 31;

  const size_t qoff = ((size_t)b * SDIM + qhalf * 128 + wave * 16) * DDIM;
  const size_t koff = (size_t)b * SDIM * DDIM;

  v8f acc[16];
#pragma unroll
  for (int t = 0; t < 16; ++t)
#pragma unroll
    for (int e = 0; e < 8; ++e) acc[t][e] = 0.0f;

  // Cooperative async stage of kh[:, k0:k0+32] into Ks[buf]:
  // 256 rows x 64 B = 1024 x 16 B chunks, 4 per thread (4 async ops per wave).
  auto stage = [&](int buf, int k0) {
    const bf16_t* src = kh + koff + k0;
#pragma unroll
    for (int i = 0; i < 4; ++i) {
      const int c    = tid + i * 256;
      const int row  = c >> 2;
      const int part = c & 3;
      async_copy_b128(src + (size_t)row * DDIM + part * 8, &Ks[buf][row][part * 8]);
    }
  };

  stage(0, 0);
  for (int step = 0; step < 24; ++step) {
    const int buf = step & 1;
    if (step < 23) {
      stage(buf ^ 1, (step + 1) * 32);                    // prefetch next slice
      asm volatile("s_wait_asynccnt 0x4" ::: "memory");   // current slice landed
    } else {
      asm volatile("s_wait_asynccnt 0x0" ::: "memory");
    }
    __syncthreads();

    const v16bf a = load_a_frag(qh + qoff + step * 32, DDIM);
#pragma unroll
    for (int t = 0; t < 16; ++t) {
      const v16bf bfrag = load_b_frag(&Ks[buf][t * 16][0], 32);
      acc[t] = wmma_bf16(a, bfrag, acc[t]);
    }
    __syncthreads();  // all waves done reading buf before it is overwritten
  }

  // Scatter accumulators to LDS: element (m, n) lives in VGPR m&7,
  // lane (n&15) + 16*(m>>3).
  const int rbase = (lane >> 4) * 8;
#pragma unroll
  for (int t = 0; t < 16; ++t)
#pragma unroll
    for (int r = 0; r < 8; ++r)
      sc[wave][rbase + r][t * 16 + (lane & 15)] = acc[t][r];
  __syncthreads();

  // 128 threads each own one q row: mask, then 8-pass max scan
  // (strict '>' ascending scan == JAX top_k lowest-index tie-break).
  if (tid < 128) {
    const int qrow = qhalf * 128 + tid;
    float* rowp = &sc[tid >> 4][tid & 15][0];
    const unsigned char* mrow = mask + ((size_t)b * SDIM + qrow) * SDIM;
    for (int j = 0; j < SDIM; ++j)
      if (!mrow[j]) rowp[j] = NEG_INF;
#pragma unroll 1
    for (int p = 0; p < KSEL; ++p) {
      float bestv = NEG_INF;
      int best = 0;
      for (int j = 0; j < SDIM; ++j) {
        const float v = rowp[j];
        if (v > bestv) { bestv = v; best = j; }
      }
      rowp[best] = NEG_INF;
      atomicAdd(&count[b * HIST + best], 1u);
    }
  }
}

// ---------------------------------------------------------------------------
// Kernel 3: top-8 of each batch histogram (ties -> lowest index), float out.
// ---------------------------------------------------------------------------
__global__ void final_topk_kernel(const unsigned int* __restrict__ count,
                                  float* __restrict__ out) {
  const int b = blockIdx.x * blockDim.x + threadIdx.x;
  if (b >= BDIM) return;
  const unsigned int* row = count + b * HIST;
  int chosen[KSEL];
#pragma unroll 1
  for (int p = 0; p < KSEL; ++p) {
    int best = 0;
    long long bestv = -1;
    for (int j = 0; j < HIST; ++j) {
      bool used = false;
#pragma unroll
      for (int q = 0; q < KSEL; ++q)
        if (q < p && chosen[q] == j) used = true;
      if (used) continue;
      const long long v = (long long)row[j];
      if (v > bestv) { bestv = v; best = j; }
    }
    chosen[p] = best;
    out[b * KSEL + p] = (float)best;
  }
}

// ---------------------------------------------------------------------------
extern "C" void kernel_launch(void* const* d_in, const int* in_sizes, int n_in,
                              void* d_out, int out_size, void* d_ws, size_t ws_size,
                              hipStream_t stream) {
  (void)in_sizes; (void)n_in; (void)out_size; (void)ws_size;

  // setup_inputs order: k_in, v_in, q_in, mask, Wk, bk, Wv, bv, Wq, bq
  const float* k_in = (const float*)d_in[0];
  // v_in (d_in[1]), Wv (d_in[6]), bv (d_in[7]) are dead code in the reference.
  const float* q_in = (const float*)d_in[2];
  const unsigned char* mask = (const unsigned char*)d_in[3];
  const float* Wk = (const float*)d_in[4];
  const float* bk = (const float*)d_in[5];
  const float* Wq = (const float*)d_in[8];
  const float* bq = (const float*)d_in[9];

  char* ws = (char*)d_ws;
  size_t off = 0;
  auto wsalloc = [&](size_t bytes) -> void* {
    void* p = ws + off;
    off += (bytes + 255) & ~(size_t)255;
    return p;
  };

  bf16_t* Whq = (bf16_t*)wsalloc((size_t)DDIM * DDIM * sizeof(bf16_t));
  bf16_t* Whk = (bf16_t*)wsalloc((size_t)DDIM * DDIM * sizeof(bf16_t));
  float*  bqs = (float*)wsalloc((size_t)DDIM * sizeof(float));
  bf16_t* qh  = (bf16_t*)wsalloc((size_t)BS * DDIM * sizeof(bf16_t));
  bf16_t* kh  = (bf16_t*)wsalloc((size_t)BS * DDIM * sizeof(bf16_t));
  unsigned int* count = (unsigned int*)wsalloc((size_t)BDIM * HIST * sizeof(unsigned int));

  prep_kernel<<<(DDIM * DDIM + 255) / 256, 256, 0, stream>>>(Wq, Wk, bq, Whq, Whk, bqs, count);
  proj_kernel<<<BS / 64, 256, 0, stream>>>(q_in, Whq, bqs, qh);
  proj_kernel<<<BS / 64, 256, 0, stream>>>(k_in, Whk, bk, kh);
  scores_topk_kernel<<<dim3(BDIM, 2), 256, 0, stream>>>(qh, kh, mask, count);
  final_topk_kernel<<<2, 256, 0, stream>>>(count, (float*)d_out);
}
